// SpatialGraphEncoder_19310172962865
// MI455X (gfx1250) — matrix operand order repaired and checked by reading, not compiled, not run on hardware
//
#include <hip/hip_runtime.h>
#include <hip/hip_bf16.h>

typedef __attribute__((ext_vector_type(16))) __bf16 v16bf;
typedef __attribute__((ext_vector_type(8)))  __bf16 bf16x8;
typedef __attribute__((ext_vector_type(8)))  float  v8f;

#define NJ 17   // real joints
#define JP 32   // padded joints (= WMMA K for the adjacency GEMM)

union FragAB { v16bf v; bf16x8 h[2]; };

// CDNA5 split-counter fence: order wave-local LDS phases (write->read reuse of
// the same per-wave buffers). "memory" clobber stops compiler reordering.
__device__ __forceinline__ void lds_fence() {
    asm volatile("s_wait_dscnt 0x0" ::: "memory");
}

__device__ __forceinline__ v8f wmma_bf16(v16bf a, v16bf b, v8f c) {
    // D = A(16x32 bf16) * B(32x16 bf16) + C(16x16 f32)
    return __builtin_amdgcn_wmma_f32_16x16x32_bf16(false, a, false, b,
                                                   (short)0, c, false, false);
}

// A fragment: 16x32 bf16, source stored K-major per row (row stride = `stride` elems).
// Lane layout (ISA 7.12.2): lanes 0-15 row=l16 hold K {kb+0..7, kb+16..23},
// lanes 16-31 hold K {kb+8..15, kb+24..31}  -> two contiguous 16B loads.
__device__ __forceinline__ v16bf load_frag_A(const __bf16* S, int row, int stride,
                                             int kb, int lo) {
    FragAB f;
    const __bf16* p = S + row * stride + kb + (lo ? 0 : 8);
    f.h[0] = *(const bf16x8*)(p);
    f.h[1] = *(const bf16x8*)(p + 16);
    return f.v;
}

// B fragment: 32x16 bf16, source stored column-major (column stride = `stride` elems,
// K contiguous down a column). Lanes 0-15 col=l16 hold K {kb+0..7, kb+16..23},
// lanes 16-31 hold K {kb+8..15, kb+24..31} -> two 16B loads.
__device__ __forceinline__ v16bf load_frag_B(const __bf16* S, int col, int stride,
                                             int kb, int lo) {
    FragAB f;
    const __bf16* p = S + col * stride + kb + (lo ? 0 : 8);
    f.h[0] = *(const bf16x8*)(p);
    f.h[1] = *(const bf16x8*)(p + 16);
    return f.v;
}

// D fragment (16x16 f32): lane holds fixed column n = l16, 8 contiguous rows
// m = m0 + (lo?0:8) + i. Store TRANSPOSED (dest[n][m...]) -> one ds_write_b128.
__device__ __forceinline__ void store_D_T(__bf16* S, int n, int stride, int m0,
                                          const v8f& d, int lo) {
    bf16x8 h;
#pragma unroll
    for (int i = 0; i < 8; ++i) h[i] = (__bf16)d[i];
    *(bf16x8*)(S + n * stride + m0 + (lo ? 0 : 8)) = h;
}

// One graph-conv layer: actOut = relu( (A @ act) @ W + b ), all via WMMA.
// act/actOut feature-major [F][JP]; mix joint-major [JP][64]; W col-major [Fout][wstride].
template <int MF_TILES, int K_TILES, int FO_TILES, bool RELU>
__device__ __forceinline__ void gc_layer(const __bf16* act, __bf16* mix,
                                         const __bf16* adjP, const __bf16* W,
                                         int wstride, const float* bias,
                                         __bf16* actOut, int lo, int l16) {
    // mix^T = act^T @ adjP   (M = features, N = joints, K = 32 joints, 1 k-step)
#pragma unroll
    for (int nj = 0; nj < 2; ++nj) {
        v16bf b = load_frag_B(adjP, nj * 16 + l16, JP, 0, lo);
#pragma unroll
        for (int mf = 0; mf < MF_TILES; ++mf) {
            v16bf a = load_frag_A(act, mf * 16 + l16, JP, 0, lo);
            v8f c = {0.f, 0.f, 0.f, 0.f, 0.f, 0.f, 0.f, 0.f};
            c = wmma_bf16(a, b, c);
            store_D_T(mix, nj * 16 + l16, 64, mf * 16, c, lo);
        }
    }
    lds_fence();
    // actOut^T: support = mix @ W  (M = joints, N = Fout, K = Fin)
#pragma unroll
    for (int nf = 0; nf < FO_TILES; ++nf) {
        float bv = bias[nf * 16 + l16];
#pragma unroll
        for (int mj = 0; mj < 2; ++mj) {
            v8f c = {0.f, 0.f, 0.f, 0.f, 0.f, 0.f, 0.f, 0.f};
#pragma unroll
            for (int kt = 0; kt < K_TILES; ++kt) {
                v16bf a = load_frag_A(mix, mj * 16 + l16, 64, kt * 32, lo);
                v16bf b = load_frag_B(W, nf * 16 + l16, wstride, kt * 32, lo);
                c = wmma_bf16(a, b, c);
            }
            v8f r;
#pragma unroll
            for (int i = 0; i < 8; ++i) {
                float v = c[i] + bv;
                r[i] = RELU ? fmaxf(v, 0.f) : v;
            }
            store_D_T(actOut, nf * 16 + l16, JP, mj * 16, r, lo);
        }
    }
    lds_fence();
}

// Final layer: same math, epilogue streams f32 to global (rows j<17 only).
template <int MF_TILES, int K_TILES, int FO_TILES>
__device__ __forceinline__ void gc_layer_last(const __bf16* act, __bf16* mix,
                                              const __bf16* adjP, const __bf16* W,
                                              int wstride, const float* bias,
                                              float* __restrict__ out, int lo, int l16) {
#pragma unroll
    for (int nj = 0; nj < 2; ++nj) {
        v16bf b = load_frag_B(adjP, nj * 16 + l16, JP, 0, lo);
#pragma unroll
        for (int mf = 0; mf < MF_TILES; ++mf) {
            v16bf a = load_frag_A(act, mf * 16 + l16, JP, 0, lo);
            v8f c = {0.f, 0.f, 0.f, 0.f, 0.f, 0.f, 0.f, 0.f};
            c = wmma_bf16(a, b, c);
            store_D_T(mix, nj * 16 + l16, 64, mf * 16, c, lo);
        }
    }
    lds_fence();
#pragma unroll
    for (int nf = 0; nf < FO_TILES; ++nf) {
        float bv = bias[nf * 16 + l16];
        int f = nf * 16 + l16;
#pragma unroll
        for (int mj = 0; mj < 2; ++mj) {
            v8f c = {0.f, 0.f, 0.f, 0.f, 0.f, 0.f, 0.f, 0.f};
#pragma unroll
            for (int kt = 0; kt < K_TILES; ++kt) {
                v16bf a = load_frag_A(mix, mj * 16 + l16, 64, kt * 32, lo);
                v16bf b = load_frag_B(W, nf * 16 + l16, wstride, kt * 32, lo);
                c = wmma_bf16(a, b, c);
            }
            int jb = mj * 16 + (lo ? 0 : 8);
#pragma unroll
            for (int i = 0; i < 8; ++i) {
                int j = jb + i;
                if (j < NJ) out[j * 32 + f] = c[i] + bv;
            }
        }
    }
}

__global__ __launch_bounds__(128, 1)
void sge_kernel(const float* __restrict__ x,
                const float* __restrict__ W1, const float* __restrict__ b1,
                const float* __restrict__ W2, const float* __restrict__ b2,
                const float* __restrict__ W3, const float* __restrict__ b3,
                const float* __restrict__ adj, float* __restrict__ out,
                int nSlices) {
    __shared__ __align__(16) __bf16 sAdj[JP * JP];      // adj+I col-major (symmetric), zero-padded
    __shared__ __align__(16) __bf16 sW1[64 * 32];       // W1^T, K padded 2->32 with zeros
    __shared__ __align__(16) __bf16 sW2[64 * 64];       // W2^T
    __shared__ __align__(16) __bf16 sW3[32 * 64];       // W3^T
    __shared__ float sB1[64], sB2[64], sB3[32];
    __shared__ __align__(16) __bf16 sAct[4][64 * JP];   // per-wave activations, feature-major
    __shared__ __align__(16) __bf16 sMix[4][JP * 64];   // per-wave A@act, joint-major

    const int t = threadIdx.x;
    const int wave = t >> 5;
    const int lane = t & 31;
    const int lo = (lane < 16);
    const int l16 = lane & 15;

    // ---- cooperative setup: stage weights/adjacency as bf16 in WMMA-friendly layouts
    for (int i = t; i < JP * JP; i += 128) {
        int c = i >> 5, k = i & 31;
        sAdj[i] = (c < NJ && k < NJ) ? (__bf16)adj[k * NJ + c] : (__bf16)0.f;
    }
    for (int i = t; i < 64 * 32; i += 128) {
        int fo = i >> 5, fi = i & 31;
        sW1[i] = (fi < 2) ? (__bf16)W1[fi * 64 + fo] : (__bf16)0.f;
    }
    for (int i = t; i < 64 * 64; i += 128) {
        int fo = i >> 6, fi = i & 63;
        sW2[i] = (__bf16)W2[fi * 64 + fo];
    }
    for (int i = t; i < 32 * 64; i += 128) {
        int fo = i >> 6, fi = i & 63;
        sW3[i] = (__bf16)W3[fi * 32 + fo];
    }
    if (t < 64) { sB1[t] = b1[t]; sB2[t] = b2[t]; }
    if (t < 32) sB3[t] = b3[t];
    {   // scrub this wave's mix buffer once (poison hygiene for layer-1 K padding)
        unsigned* mz = (unsigned*)&sMix[wave][0];
        for (int i = lane; i < (JP * 64) / 2; i += 32) mz[i] = 0u;
    }
    __syncthreads();

    __bf16* act = &sAct[wave][0];
    __bf16* mix = &sMix[wave][0];
    const int stride = gridDim.x * 4;

    for (int slice = blockIdx.x * 4 + wave; slice < nSlices; slice += stride) {
        const float* xs = x + (long)slice * (NJ * 2);
        {   // prefetch the slice this wave will touch next iteration
            int ns = slice + stride;
            if (ns < nSlices) __builtin_prefetch(x + (long)ns * (NJ * 2), 0, 0);
        }
        // x (17x2 f32, row-major) -> act feature-major rows 0..15 (pads zeroed)
        {
            unsigned* az = (unsigned*)act;
#pragma unroll
            for (int i = lane; i < 256; i += 32) az[i] = 0u;  // rows 0..15 x 32 cols
#pragma unroll
            for (int r = 0; r < 2; ++r) {
                int idx = lane + 32 * r;
                if (idx < NJ * 2) {
                    int j = idx >> 1, f = idx & 1;
                    act[f * JP + j] = (__bf16)xs[idx];
                }
            }
        }
        lds_fence();
        gc_layer<1, 1, 4, true>(act, mix, sAdj, sW1, 32, sB1, act, lo, l16);
        gc_layer<4, 2, 4, true>(act, mix, sAdj, sW2, 64, sB2, act, lo, l16);
        gc_layer_last<4, 2, 2>(act, mix, sAdj, sW3, 64, sB3,
                               out + (long)slice * (NJ * 32), lo, l16);
    }
}

extern "C" void kernel_launch(void* const* d_in, const int* in_sizes, int n_in,
                              void* d_out, int out_size, void* d_ws, size_t ws_size,
                              hipStream_t stream) {
    const float* x   = (const float*)d_in[0];
    const float* W1  = (const float*)d_in[1];
    const float* b1  = (const float*)d_in[2];
    const float* W2  = (const float*)d_in[3];
    const float* b2  = (const float*)d_in[4];
    const float* W3  = (const float*)d_in[5];
    const float* b3  = (const float*)d_in[6];
    const float* adj = (const float*)d_in[7];
    float* out = (float*)d_out;

    int nSlices = in_sizes[0] / (NJ * 2);   // B*T = 65536
    int grid = 2048;                         // 8192 waves -> 8 slices/wave
    int maxg = (nSlices + 3) / 4;
    if (grid > maxg) grid = maxg;

    sge_kernel<<<dim3(grid), dim3(128), 0, stream>>>(x, W1, b1, W2, b2, W3, b3,
                                                     adj, out, nSlices);
}
